// GraphLayer_51367808860367
// MI455X (gfx1250) — matrix operand order repaired and checked by reading, not compiled, not run on hardware
//
#include <hip/hip_runtime.h>

// ---------------------------------------------------------------------------
// Bipartite GNN layer for MI455X (gfx1250, wave32, WMMA bf16 -> f32 acc).
//
// Pipeline per conv:
//   Pr = right @ Wl^T + bl          (node GEMM, f32 out in ws)
//   Pl = left  @ Wr^T               (node GEMM, f32 out in ws)
//   agg = 0
//   edge kernel: pre = edge@We^T + Pr[dst] + Pl[src]; LN; leaky; @Wf^T; +bf;
//                atomicAdd into agg[dst]
//   node out: LN2(agg) || right -> leaky(@Wo1^T+bo1) -> leaky(@Wo2^T+bo2)
//
// Workspace layout (needs ~32 MB):
//   [0, 768KB)    : 12 bf16 weight matrices, 64KB slots
//   Pr / Pl / agg : 3 x 20000*128*4 bytes
// ---------------------------------------------------------------------------

#define D 128
#define LDA 136   // padded bf16 row stride for 128-wide tiles (conflict-free)
#define LDH 264   // padded bf16 row stride for 256-wide tiles

typedef __attribute__((ext_vector_type(8)))  __bf16 v8bf;
typedef __attribute__((ext_vector_type(16))) __bf16 v16bf;
typedef __attribute__((ext_vector_type(8)))  float  v8f;

static __device__ __forceinline__ v16bf cat8(v8bf lo, v8bf hi) {
  v16bf r;
#pragma unroll
  for (int i = 0; i < 8; ++i) { r[i] = lo[i]; r[i + 8] = hi[i]; }
  return r;
}

// A operand: 16x32 bf16 tile, row-major source with stride `lda`.
// Lane layout (ISA 7.12.2): M = lane&15; half = lane>>4;
//   v0..3 = K  (k0 + half*8 .. +7), v4..7 = K (k0 + 16 + half*8 .. +7)
static __device__ __forceinline__ v16bf load_a(const __bf16* base, int lda,
                                               int m, int half, int k0) {
  const __bf16* p = base + m * lda + k0 + half * 8;
  v8bf lo = *(const v8bf*)p;
  v8bf hi = *(const v8bf*)(p + 16);
  return cat8(lo, hi);
}

// B operand: 32x16 bf16 tile of W^T, i.e. B(k,n) = W[n*ldw + k].
// Lane layout: N = lane&15; v0..3 = K (k0 + half*8 .. +7), v4..7 = K (k0 + 16 + half*8 .. +7).
static __device__ __forceinline__ v16bf load_b(const __bf16* W, int ldw,
                                               int n, int k0, int half) {
  const __bf16* p = W + (size_t)n * ldw + k0 + half * 8;
  v8bf lo = *(const v8bf*)p;
  v8bf hi = *(const v8bf*)(p + 16);
  return cat8(lo, hi);
}

static __device__ __forceinline__ v8f wmma_bf16(v16bf a, v16bf b, v8f c) {
  return __builtin_amdgcn_wmma_f32_16x16x32_bf16(false, a, false, b,
                                                 (short)0, c, false, false);
}

static __device__ __forceinline__ float leaky(float x) {
  return x >= 0.f ? x : 0.01f * x;
}

// ---------------------------------------------------------------------------
__global__ __launch_bounds__(256) void cvt_bf16_kernel(
    const float* __restrict__ in, __bf16* __restrict__ out, int n) {
  int i = blockIdx.x * 256 + threadIdx.x;
  if (i < n) out[i] = (__bf16)in[i];
}

__global__ __launch_bounds__(256) void zero_kernel(float* __restrict__ p, int n) {
  int i = blockIdx.x * 256 + threadIdx.x;
  if (i < n) p[i] = 0.f;
}

// ---------------------------------------------------------------------------
// out[r, :] = X[r, :] @ W^T (+ bias).   One wave = 16 rows.
__global__ __launch_bounds__(256) void proj_kernel(
    const float* __restrict__ X, const __bf16* __restrict__ W,
    const float* __restrict__ bias, float* __restrict__ out, int Nn) {
  __shared__ __align__(16) __bf16 lds[8][16 * LDA];
  const int lane = threadIdx.x & 31, w = threadIdx.x >> 5;
  const int tile = blockIdx.x * 8 + w;
  const int r0 = tile * 16;
  const bool valid = r0 < Nn;
  __bf16* myl = &lds[w][0];

  if (valid) {
    for (int i = lane; i < 16 * D; i += 32) {
      int r = i >> 7, c = i & 127;
      myl[r * LDA + c] = (__bf16)X[(size_t)(r0 + r) * D + c];
    }
  }
  __syncthreads();

  if (valid) {
    const int half = lane >> 4, nlo = lane & 15;
    v8f acc[8] = {};
#pragma unroll
    for (int ks = 0; ks < 4; ++ks) {
      v16bf a = load_a(myl, LDA, nlo, half, ks * 32);
#pragma unroll
      for (int nt = 0; nt < 8; ++nt) {
        v16bf b = load_b(W, D, nt * 16 + nlo, ks * 32, half);
        acc[nt] = wmma_bf16(a, b, acc[nt]);
      }
    }
#pragma unroll
    for (int nt = 0; nt < 8; ++nt) {
      int N = nt * 16 + nlo;
      float bb = bias ? bias[N] : 0.f;
#pragma unroll
      for (int r = 0; r < 8; ++r) {
        out[(size_t)(r0 + r + 8 * half) * D + N] = acc[nt][r] + bb;
      }
    }
  }
}

// ---------------------------------------------------------------------------
// Edge message kernel.  One wave = 16 edges (full 128-wide output).
__global__ __launch_bounds__(256) void edge_kernel(
    const float* __restrict__ Pr, const float* __restrict__ Pl,
    const float* __restrict__ ef,
    const int* __restrict__ src, const int* __restrict__ dst,
    const __bf16* __restrict__ We, const __bf16* __restrict__ Wf,
    const float* __restrict__ g1, const float* __restrict__ b1,
    const float* __restrict__ bfb,
    float* __restrict__ agg, int E) {
  __shared__ __align__(16) __bf16 lds[8][16 * LDA];
  const int lane = threadIdx.x & 31, w = threadIdx.x >> 5;
  const int tile = blockIdx.x * 8 + w;
  const int e0 = tile * 16;
  const bool valid = e0 < E;
  __bf16* myl = &lds[w][0];
  const int half = lane >> 4, nlo = lane & 15;

  int dstm[8];
  float bfv[8];
  v8f acc[8] = {};

  // Stage edge-feature tile as bf16 (coalesced f32 reads).
  if (valid) {
    for (int i = lane; i < 16 * D; i += 32) {
      int r = i >> 7, c = i & 127;
      myl[r * LDA + c] = (__bf16)ef[(size_t)(e0 + r) * D + c];
    }
  }
  __syncthreads();

  if (valid) {
    // GEMM1: pre = edge @ We^T
#pragma unroll
    for (int ks = 0; ks < 4; ++ks) {
      v16bf a = load_a(myl, LDA, nlo, half, ks * 32);
#pragma unroll
      for (int nt = 0; nt < 8; ++nt) {
        v16bf b = load_b(We, D, nt * 16 + nlo, ks * 32, half);
        acc[nt] = wmma_bf16(a, b, acc[nt]);
      }
    }

    // + Pr[dst] + Pl[src]   (C layout: row M = r + 8*half, col N = nt*16+nlo)
    int srcm[8];
#pragma unroll
    for (int r = 0; r < 8; ++r) {
      dstm[r] = dst[e0 + r + 8 * half];
      srcm[r] = src[e0 + r + 8 * half];
    }
    float gv[8], bv[8];
#pragma unroll
    for (int nt = 0; nt < 8; ++nt) {
      int N = nt * 16 + nlo;
      gv[nt] = g1[N]; bv[nt] = b1[N]; bfv[nt] = bfb[N];
    }
#pragma unroll
    for (int nt = 0; nt < 8; ++nt) {
      int N = nt * 16 + nlo;
#pragma unroll
      for (int r = 0; r < 8; ++r) {
        acc[nt][r] += Pr[(size_t)dstm[r] * D + N] + Pl[(size_t)srcm[r] * D + N];
      }
    }

    // Row LayerNorm across the 16 lanes of each half, then LeakyReLU.
#pragma unroll
    for (int r = 0; r < 8; ++r) {
      float s = 0.f, q = 0.f;
#pragma unroll
      for (int nt = 0; nt < 8; ++nt) { float v = acc[nt][r]; s += v; q += v * v; }
#pragma unroll
      for (int m = 1; m < 16; m <<= 1) {
        s += __shfl_xor(s, m, 32);
        q += __shfl_xor(q, m, 32);
      }
      float mu = s * 0.0078125f;
      float rs = rsqrtf(fmaxf(q * 0.0078125f - mu * mu, 0.f) + 1e-5f);
#pragma unroll
      for (int nt = 0; nt < 8; ++nt) {
        acc[nt][r] = leaky((acc[nt][r] - mu) * rs * gv[nt] + bv[nt]);
      }
    }
  }
  __syncthreads();

  // Restage activation as bf16 A tile.
  if (valid) {
#pragma unroll
    for (int nt = 0; nt < 8; ++nt) {
      int N = nt * 16 + nlo;
#pragma unroll
      for (int r = 0; r < 8; ++r) myl[(r + 8 * half) * LDA + N] = (__bf16)acc[nt][r];
    }
  }
  __syncthreads();

  if (valid) {
    // GEMM2: msg = act @ Wf^T ; then += bf ; atomic scatter-add into agg[dst].
    v8f acc2[8] = {};
#pragma unroll
    for (int ks = 0; ks < 4; ++ks) {
      v16bf a = load_a(myl, LDA, nlo, half, ks * 32);
#pragma unroll
      for (int nt = 0; nt < 8; ++nt) {
        v16bf b = load_b(Wf, D, nt * 16 + nlo, ks * 32, half);
        acc2[nt] = wmma_bf16(a, b, acc2[nt]);
      }
    }
#pragma unroll
    for (int nt = 0; nt < 8; ++nt) {
      int N = nt * 16 + nlo;
#pragma unroll
      for (int r = 0; r < 8; ++r) {
        atomicAdd(&agg[(size_t)dstm[r] * D + N], acc2[nt][r] + bfv[nt]);
      }
    }
  }
}

// ---------------------------------------------------------------------------
// Output head: LN2(agg) concat right -> leaky(@Wo1^T+bo1) -> leaky(@Wo2^T+bo2)
// One wave = 16 rows; 4 waves / workgroup (LDS budget).
__global__ __launch_bounds__(128) void nodeout_kernel(
    const float* __restrict__ agg, const float* __restrict__ right,
    const float* __restrict__ g2, const float* __restrict__ b2,
    const __bf16* __restrict__ Wo1, const float* __restrict__ bo1,
    const __bf16* __restrict__ Wo2, const float* __restrict__ bo2,
    float* __restrict__ out, int Nn) {
  __shared__ __align__(16) __bf16 ldsH[4][16 * LDH];
  __shared__ __align__(16) __bf16 ldsA[4][16 * LDA];
  __shared__ float stats[4][32];
  const int lane = threadIdx.x & 31, w = threadIdx.x >> 5;
  const int tile = blockIdx.x * 4 + w;
  const int r0 = tile * 16;
  const bool valid = r0 < Nn;
  const int half = lane >> 4, nlo = lane & 15;

  // LayerNorm stats: 2 lanes per row (each sums 64 cols), combine via shfl.
  if (valid) {
    const float* ar = agg + (size_t)(r0 + nlo) * D + half * 64;
    float s = 0.f, q = 0.f;
    for (int c = 0; c < 64; ++c) { float v = ar[c]; s += v; q += v * v; }
    s += __shfl_xor(s, 16, 32);
    q += __shfl_xor(q, 16, 32);
    float mu = s * 0.0078125f;
    float rs = rsqrtf(fmaxf(q * 0.0078125f - mu * mu, 0.f) + 1e-5f);
    if (half == 0) { stats[w][nlo * 2] = mu; stats[w][nlo * 2 + 1] = rs; }
  }
  __syncthreads();

  // Build h = [LN(agg), right] as a 16x256 bf16 tile.
  if (valid) {
    for (int i = lane; i < 16 * 256; i += 32) {
      int r = i >> 8, c = i & 255;
      float v;
      if (c < D) {
        v = (agg[(size_t)(r0 + r) * D + c] - stats[w][r * 2]) *
                stats[w][r * 2 + 1] * g2[c] + b2[c];
      } else {
        v = right[(size_t)(r0 + r) * D + (c - D)];
      }
      ldsH[w][r * LDH + c] = (__bf16)v;
    }
  }
  __syncthreads();

  // GEMM1: K = 256.
  if (valid) {
    v8f acc[8] = {};
#pragma unroll
    for (int ks = 0; ks < 8; ++ks) {
      v16bf a = load_a(&ldsH[w][0], LDH, nlo, half, ks * 32);
#pragma unroll
      for (int nt = 0; nt < 8; ++nt) {
        v16bf b = load_b(Wo1, 256, nt * 16 + nlo, ks * 32, half);
        acc[nt] = wmma_bf16(a, b, acc[nt]);
      }
    }
#pragma unroll
    for (int nt = 0; nt < 8; ++nt) {
      int N = nt * 16 + nlo;
      float bb = bo1[N];
#pragma unroll
      for (int r = 0; r < 8; ++r) {
        ldsA[w][(r + 8 * half) * LDA + N] = (__bf16)leaky(acc[nt][r] + bb);
      }
    }
  }
  __syncthreads();

  // GEMM2: K = 128, write final f32 output.
  if (valid) {
    v8f a2[8] = {};
#pragma unroll
    for (int ks = 0; ks < 4; ++ks) {
      v16bf a = load_a(&ldsA[w][0], LDA, nlo, half, ks * 32);
#pragma unroll
      for (int nt = 0; nt < 8; ++nt) {
        v16bf b = load_b(Wo2, D, nt * 16 + nlo, ks * 32, half);
        a2[nt] = wmma_bf16(a, b, a2[nt]);
      }
    }
#pragma unroll
    for (int nt = 0; nt < 8; ++nt) {
      int N = nt * 16 + nlo;
      float bb = bo2[N];
#pragma unroll
      for (int r = 0; r < 8; ++r) {
        out[(size_t)(r0 + r + 8 * half) * D + N] = leaky(a2[nt][r] + bb);
      }
    }
  }
}

// ---------------------------------------------------------------------------
extern "C" void kernel_launch(void* const* d_in, const int* in_sizes, int n_in,
                              void* d_out, int out_size, void* d_ws, size_t ws_size,
                              hipStream_t stream) {
  const float* items = (const float*)d_in[0];
  const int*   eidx  = (const int*)d_in[1];
  const float* ef    = (const float*)d_in[2];
  const float* boxes = (const float*)d_in[3];

  const int Nn = in_sizes[0] / D;        // 20000
  const int E  = in_sizes[1] / 2;        // 640000
  const int* src = eidx;
  const int* dst = eidx + E;

  // Per-conv param indices: base+0 Wl, +1 bl, +2 We, +3 Wr, +4 ln1_g, +5 ln1_b,
  // +6 Wf, +7 bf, +8 ln2_g, +9 ln2_b, +10 Wo1, +11 bo1, +12 Wo2, +13 bo2.
  auto P = [&](int conv, int j) { return (const float*)d_in[4 + conv * 14 + j]; };

  // Workspace layout.
  char* ws = (char*)d_ws;
  auto Wslot = [&](int conv, int s) {            // 12 slots of 64KB
    return (__bf16*)(ws + (size_t)(conv * 6 + s) * 65536);
  };
  const size_t PROJ = (size_t)Nn * D * sizeof(float);
  float* Pr  = (float*)(ws + (12 * 65536));
  float* Pl  = (float*)(ws + (12 * 65536) + PROJ);
  float* agg = (float*)(ws + (12 * 65536) + 2 * PROJ);

  float* out_items = (float*)d_out;
  float* out_boxes = out_items + (size_t)Nn * D;

  // --- Convert all weight matrices to bf16 -------------------------------
  const int wsrc[6] = {0, 2, 3, 6, 10, 12};       // Wl, We, Wr, Wf, Wo1, Wo2
  for (int c = 0; c < 2; ++c) {
    for (int s = 0; s < 6; ++s) {
      int n = (s == 4) ? D * 256 : D * D;
      cvt_bf16_kernel<<<(n + 255) / 256, 256, 0, stream>>>(P(c, wsrc[s]),
                                                           Wslot(c, s), n);
    }
  }

  const int ntile   = (Nn + 15) / 16;
  dim3 gProj((ntile + 7) / 8);
  dim3 gEdge(((E + 15) / 16 + 7) / 8);
  dim3 gOut((ntile + 3) / 4);
  dim3 gZero(((Nn * D) + 255) / 256);

  // ---------------- conv1: items --(src->dst)--> boxes -------------------
  proj_kernel<<<gProj, 256, 0, stream>>>(boxes, Wslot(0, 0), P(0, 1), Pr, Nn);
  proj_kernel<<<gProj, 256, 0, stream>>>(items, Wslot(0, 2), nullptr, Pl, Nn);
  zero_kernel<<<gZero, 256, 0, stream>>>(agg, Nn * D);
  edge_kernel<<<gEdge, 256, 0, stream>>>(Pr, Pl, ef, src, dst,
                                         Wslot(0, 1), Wslot(0, 3),
                                         P(0, 4), P(0, 5), P(0, 7), agg, E);
  nodeout_kernel<<<gOut, 128, 0, stream>>>(agg, boxes, P(0, 8), P(0, 9),
                                           Wslot(0, 4), P(0, 11),
                                           Wslot(0, 5), P(0, 13),
                                           out_boxes, Nn);

  // ---------------- conv2: proc_boxes --(dst->src)--> items --------------
  proj_kernel<<<gProj, 256, 0, stream>>>(items, Wslot(1, 0), P(1, 1), Pr, Nn);
  proj_kernel<<<gProj, 256, 0, stream>>>(out_boxes, Wslot(1, 2), nullptr, Pl, Nn);
  zero_kernel<<<gZero, 256, 0, stream>>>(agg, Nn * D);
  edge_kernel<<<gEdge, 256, 0, stream>>>(Pr, Pl, ef, /*src=*/dst, /*dst=*/src,
                                         Wslot(1, 1), Wslot(1, 3),
                                         P(1, 4), P(1, 5), P(1, 7), agg, E);
  nodeout_kernel<<<gOut, 128, 0, stream>>>(agg, items, P(1, 8), P(1, 9),
                                           Wslot(1, 4), P(1, 11),
                                           Wslot(1, 5), P(1, 13),
                                           out_items, Nn);
}
